// MultiScaleVQVAE_19851338842615
// MI455X (gfx1250) — matrix operand order repaired and checked by reading, not compiled, not run on hardware
//
#include <hip/hip_runtime.h>
#include <math.h>

typedef __attribute__((ext_vector_type(16))) _Float16 v16h;
typedef __attribute__((ext_vector_type(8)))  _Float16 v8h;
typedef __attribute__((ext_vector_type(8)))  float    v8f;
typedef __attribute__((ext_vector_type(4))) unsigned int u32x4;
typedef __attribute__((ext_vector_type(8))) int i32x8;
typedef __attribute__((ext_vector_type(4))) int i32x4;

// ---- Tensor Data Mover availability (arity differs across toolchains) ------
#if defined(__has_builtin)
#if __has_builtin(__builtin_amdgcn_tensor_load_to_lds)
#define HAVE_TDM 1
#if __has_include(<hip/amd_detail/amd_gfx1250_TDM.h>)
#warning "CDNA5 probe: TDM path ENABLED (6-arg therock form)"
#define TDM_LOAD(g0, g1, g2, g3) \
  __builtin_amdgcn_tensor_load_to_lds((g0), (g1), (g2), (g3), (i32x8){0, 0, 0, 0, 0, 0, 0, 0}, 0)
#else
#warning "CDNA5 probe: TDM path ENABLED (5-arg ROCm form)"
#define TDM_LOAD(g0, g1, g2, g3) \
  __builtin_amdgcn_tensor_load_to_lds((g0), (g1), (g2), (g3), 0)
#endif
#else
#warning "CDNA5 probe: TDM builtin NOT available - using LDS fallback"
#endif
#endif

// ---------------------------------------------------------------------------
// WMMA fragment loaders (contiguous k = 16*hi + s order, wave32).
// A 16x32 f16 row-major in LDS: per lane two contiguous 16B runs -> 2x b128.
// B fragments come from PRE-SWIZZLED storage: each lane's 16 halves contiguous.
// ---------------------------------------------------------------------------
__device__ inline v16h concat8(v8h a, v8h b) {
  return __builtin_shufflevector(a, b, 0, 1, 2, 3, 4, 5, 6, 7, 8, 9, 10, 11, 12,
                                 13, 14, 15);
}

__device__ inline v16h load_a_frag128(const _Float16* A, int ld, int k0, int lane) {
  const int m = lane & 15, hi = (lane >> 4) & 1;
  const _Float16* p = A + m * ld + k0 + hi * 16;
  v8h lo = *(const v8h*)(p);       // K = k0 + hi*16 .. +7
  v8h hh = *(const v8h*)(p + 8);   // K = k0 + hi*16 + 8 .. +15
  return concat8(lo, hh);
}

__device__ inline v16h load_b_frag128(const _Float16* Bs, int lane) {
  const _Float16* p = Bs + lane * 16;
  v8h lo = *(const v8h*)(p);
  v8h hh = *(const v8h*)(p + 8);
  return concat8(lo, hh);
}

// swizzled B staging position for element (k in 0..31, n in 0..15)
__device__ inline int b_swz(int k, int n) {
  return (n + (k & 16)) * 16 + (k & 15);
}

// ---------------------------------------------------------------------------
// One-shot weight repack: f32 (conv OIHW / convT IOHW-flipped) -> f16 packed
// fragment-ready layout bp[ct][kc][lane][16], so conv waves fetch B fragments
// as two coalesced global b128 loads. Runs once per conv, output lives in L2.
// ---------------------------------------------------------------------------
template <int Ci, int Co, int Kk, int MODE>
__global__ void repack_weights(const float* __restrict__ w,
                               _Float16* __restrict__ bp) {
  constexpr int KK = Kk * Kk;
  constexpr int Ktot = Ci * KK;
  constexpr int NKC = (Ktot + 31) / 32;
  constexpr int NCTP = ((Co + 63) / 64) * 4;  // wave-tile count incl. padding
  constexpr int total = NCTP * NKC * 512;
  const int t = blockIdx.x * blockDim.x + threadIdx.x;
  if (t >= total) return;
  const int j = t & 15;
  const int lane = (t >> 4) & 31;
  const int rest = t >> 9;
  const int kc = rest % NKC;
  const int ct = rest / NKC;
  const int n = lane & 15, hi = lane >> 4;
  const int co = ct * 16 + n;
  const int k = kc * 32 + hi * 16 + j;  // element j <-> K = j + hi*16 in chunk
  float v = 0.f;
  if (co < Co && k < Ktot) {
    const int ci = k / KK;
    const int r2 = k - ci * KK;
    const int ky = r2 / Kk, kx = r2 - ky * Kk;
    if (MODE == 0)
      v = w[((co * Ci + ci) * Kk + ky) * Kk + kx];
    else
      v = w[((ci * Co + co) * Kk + (Kk - 1 - ky)) * Kk + (Kk - 1 - kx)];
  }
  bp[t] = (_Float16)v;
}

// ---------------------------------------------------------------------------
// Implicit-GEMM conv / transposed-conv, templated geometry, 32-row M tile,
// 2 accumulators per wave. B fragments loaded straight from packed weights.
// ---------------------------------------------------------------------------
#define CONV_WAVES 4

template <int Ci, int Hi, int Wi, int Co, int Ho, int Wo, int Kk, int S, int P,
          int MODE, int RELU, int ACCUM>
__global__ __launch_bounds__(128) void conv2d_wmma(
    const float* __restrict__ in, const _Float16* __restrict__ bp,
    const float* __restrict__ bias, float* __restrict__ out) {
  constexpr int NB = 64;
  constexpr int M = NB * Ho * Wo;
  constexpr int HoWo = Ho * Wo;
  constexpr int KK = Kk * Kk;
  constexpr int Ktot = Ci * KK;
  constexpr int NKC = (Ktot + 31) / 32;

  const int lane = threadIdx.x & 31;
  const int wave = threadIdx.x >> 5;
  const int m0 = blockIdx.x * 32;
  const int ct = blockIdx.y * CONV_WAVES + wave;
  const int co0 = ct * 16;

  __shared__ __align__(16) _Float16 Atile[32 * 32];

  v8f acc0 = {}, acc1 = {};

  for (int kc = 0; kc < NKC; ++kc) {
    const int k0 = kc * 32;
    // --- cooperative A stage (row-major 32x32)
    for (int t = threadIdx.x; t < 32 * 32; t += 128) {
      const int mi = t >> 5, ki = t & 31;
      const int m = m0 + mi;
      const int kk = k0 + ki;
      float v = 0.f;
      if (m < M && kk < Ktot) {
        const int n = m / HoWo;
        const int rem = m - n * HoWo;
        const int oy = rem / Wo, ox = rem - oy * Wo;
        const int ci = kk / KK;
        const int r2 = kk - ci * KK;
        const int ky = r2 / Kk, kx = r2 - ky * Kk;
        if (MODE == 0) {
          const int iy = oy * S - P + ky;
          const int ix = ox * S - P + kx;
          if (iy >= 0 && iy < Hi && ix >= 0 && ix < Wi)
            v = in[((n * Ci + ci) * Hi + iy) * Wi + ix];
        } else {
          const int ty = oy + P - ky, tx = ox + P - kx;
          if (ty >= 0 && tx >= 0 && (ty % S) == 0 && (tx % S) == 0) {
            const int iy = ty / S, ix = tx / S;
            if (iy < Hi && ix < Wi)
              v = in[((n * Ci + ci) * Hi + iy) * Wi + ix];
          }
        }
      }
      Atile[t] = (_Float16)v;
    }
    // --- B fragment: two coalesced global b128 loads from packed weights
    const _Float16* bpp = bp + (((size_t)ct * NKC + kc) * 32 + lane) * 16;
    v8h b0 = *(const v8h*)bpp;
    v8h b1 = *(const v8h*)(bpp + 8);
    v16h b = concat8(b0, b1);
    if (kc + 1 < NKC)  // next packed chunk toward L2 (global_prefetch_b8)
      __builtin_prefetch(bp + ((size_t)ct * NKC + kc + 1) * 512, 0, 1);
    __syncthreads();
    v16h aLo = load_a_frag128(Atile, 32, 0, lane);
    v16h aHi = load_a_frag128(Atile + 16 * 32, 32, 0, lane);
    acc0 = __builtin_amdgcn_wmma_f32_16x16x32_f16(false, aLo, false, b, (short)0,
                                                  acc0, false, false);
    acc1 = __builtin_amdgcn_wmma_f32_16x16x32_f16(false, aHi, false, b, (short)0,
                                                  acc1, false, false);
    __syncthreads();
  }

  // --- epilogue: C/D row m = r + 8*hi, col n = lane&15
  const int n = lane & 15;
  const int hi = lane >> 4;
  const int co = co0 + n;
  if (co < Co) {
    const float bv = bias[co];
#pragma unroll
    for (int half = 0; half < 2; ++half) {
      const v8f& acc = half ? acc1 : acc0;
      const int mb = m0 + half * 16;
#pragma unroll
      for (int r = 0; r < 8; ++r) {
        const int m = mb + r + 8 * hi;
        if (m < M) {
          const int nn = m / HoWo;
          const int rem = m - nn * HoWo;
          const int oy = rem / Wo, ox = rem - oy * Wo;
          float v = acc[r] + bv;
          if (RELU) v = v > 0.f ? v : 0.f;
          const size_t o = ((size_t)(nn * Co + co) * Ho + oy) * Wo + ox;
          out[o] = ACCUM ? (out[o] + v) : v;
        }
      }
    }
  }
}

// ---------------------------------------------------------------------------
// VQ nearest-code search via WMMA on an f16 codebook: dist=|z|^2+|c|^2-2 z.c
// block = 256 (8 waves); wave owns one 16-row tile vs all 1024 codes.
// Codebook tiles staged by the Tensor Data Mover when available.
// ---------------------------------------------------------------------------
template <int S>
__global__ __launch_bounds__(256) void vq_argmin_wmma(
    const float* __restrict__ Fs, const _Float16* __restrict__ cbh,
    const float* __restrict__ cnorm, int* __restrict__ idxOut) {
  constexpr int Npos = 64 * S * S;
  constexpr int SS = S * S;
  const int lane = threadIdx.x & 31;
  const int wave = threadIdx.x >> 5;
  const int m0 = (blockIdx.x * 8 + wave) * 16;

  __shared__ __align__(16) _Float16 Acache[8][16 * 64];
  __shared__ float Znorm[8][16];
  __shared__ __align__(16) _Float16 Bsw[2][32 * 16];
#if defined(HAVE_TDM)
  __shared__ __align__(16) _Float16 Braw[16 * 64];  // row-major TDM landing pad
#endif

  // stage this wave's 16x64 activation tile (row-major, ld=64)
  for (int t = lane; t < 16 * 64; t += 32) {
    const int mi = t >> 6, ki = t & 63;
    const int m = m0 + mi;
    float v = 0.f;
    if (m < Npos) {
      const int n = m / SS;
      const int rem = m - n * SS;
      const int y = rem / S, x = rem - y * S;
      v = Fs[((n * 64 + ki) * S + y) * S + x];
    }
    Acache[wave][t] = (_Float16)v;
  }
  if (lane < 16) {
    float s = 0.f;
#pragma unroll
    for (int k = 0; k < 64; ++k) {
      const float v = (float)Acache[wave][lane * 64 + k];
      s += v * v;
    }
    Znorm[wave][lane] = s;
  }

  const int n = lane & 15;
  const int hi = lane >> 4;

  v16h a0 = load_a_frag128(Acache[wave], 64, 0, lane);
  v16h a1 = load_a_frag128(Acache[wave], 64, 32, lane);
  float zn[8];
#pragma unroll
  for (int r = 0; r < 8; ++r) zn[r] = Znorm[wave][r + 8 * hi];

  float best[8];
  int bidx[8];
#pragma unroll
  for (int r = 0; r < 8; ++r) { best[r] = 3.4e38f; bidx[r] = 0; }

  for (int ctile = 0; ctile < 64; ++ctile) {
    const int c0 = ctile * 16;
    __syncthreads();
#if defined(HAVE_TDM)
    // TDM: async DMA of a 16x64 f16 tile (rows contiguous, row stride 64)
    if (wave == 0) {
      const unsigned long long ga = (unsigned long long)(const void*)(cbh + c0 * 64);
      u32x4 g0;
      g0[0] = 1u;                                         // count=1, user desc
      g0[1] = (unsigned)(size_t)&Braw[0];                 // lds_addr
      g0[2] = (unsigned)(ga & 0xffffffffu);               // global_addr lo
      g0[3] = (unsigned)((ga >> 32) & 0x01ffffffu) | 0x80000000u;  // hi | type=2
      i32x8 g1;
      g1[0] = 0x00010000;  // data_size = 2 bytes
      g1[1] = 64 << 16;    // tensor_dim0 = 64
      g1[2] = 16 << 16;    // tensor_dim1 = 16
      g1[3] = 64 << 16;    // tile_dim0 = 64
      g1[4] = 16;          // tile_dim1 = 16
      g1[5] = 64;          // tensor_dim0_stride = 64
      g1[6] = 0;
      g1[7] = 0;
      const i32x4 z4 = {0, 0, 0, 0};
      TDM_LOAD(g0, g1, z4, z4);
      __builtin_amdgcn_s_wait_tensorcnt(0);
    }
    __syncthreads();
    // LDS->LDS swizzle into fragment-friendly layout
    for (int t = threadIdx.x; t < 16 * 64; t += 256) {
      const int ni = t >> 6, ki = t & 63;
      Bsw[ki >> 5][b_swz(ki & 31, ni)] = Braw[ni * 64 + ki];
    }
#else
    for (int t = threadIdx.x; t < 64 * 16; t += 256) {
      const int ki = t >> 4, ni = t & 15;
      Bsw[ki >> 5][b_swz(ki & 31, ni)] = cbh[(c0 + ni) * 64 + ki];
    }
#endif
    __syncthreads();
    v16h b0 = load_b_frag128(Bsw[0], lane);
    v16h b1 = load_b_frag128(Bsw[1], lane);
    v8f dot = {};
    dot = __builtin_amdgcn_wmma_f32_16x16x32_f16(false, a0, false, b0, (short)0,
                                                 dot, false, false);
    dot = __builtin_amdgcn_wmma_f32_16x16x32_f16(false, a1, false, b1, (short)0,
                                                 dot, false, false);
    const float cn = cnorm[c0 + n];
#pragma unroll
    for (int r = 0; r < 8; ++r) {
      const float dist = zn[r] + cn - 2.f * dot[r];
      if (dist < best[r]) { best[r] = dist; bidx[r] = c0 + n; }
    }
  }

  // reduce over the 16 candidate lanes of each half-wave
#pragma unroll
  for (int r = 0; r < 8; ++r) {
    float bd = best[r];
    int bi = bidx[r];
#pragma unroll
    for (int msk = 1; msk <= 8; msk <<= 1) {
      const float od = __shfl_xor(bd, msk, 32);
      const int oi = __shfl_xor(bi, msk, 32);
      if (od < bd || (od == bd && oi < bi)) { bd = od; bi = oi; }
    }
    if (n == 0) {
      const int mm = m0 + r + 8 * hi;
      if (mm < Npos) idxOut[mm] = bi;
    }
  }
}

// ---------------------------------------------------------------------------
// helpers
// ---------------------------------------------------------------------------
__global__ void cast_cb_f16(const float* __restrict__ cb,
                            _Float16* __restrict__ cbh) {
  const int t = blockIdx.x * blockDim.x + threadIdx.x;
  if (t < 1024 * 64) cbh[t] = (_Float16)cb[t];
}

__global__ void codebook_norms(const float* __restrict__ cb, float* __restrict__ cn) {
  const int c = blockIdx.x * blockDim.x + threadIdx.x;
  if (c < 1024) {
    float s = 0.f;
#pragma unroll 8
    for (int k = 0; k < 64; ++k) { const float v = cb[c * 64 + k]; s += v * v; }
    cn[c] = s;
  }
}

__global__ void zero_f(float* p, int n) {
  const int t = blockIdx.x * blockDim.x + threadIdx.x;
  if (t < n) p[t] = 0.f;
}
__global__ void zero_i(int* p, int n) {
  const int t = blockIdx.x * blockDim.x + threadIdx.x;
  if (t < n) p[t] = 0;
}

// jax.image.resize 'bilinear' (half-pixel centers, antialias=False)
template <int N, int C, int Hi, int Wi, int Ho, int Wo>
__global__ void bilinear_resize(const float* __restrict__ in,
                                float* __restrict__ out) {
  constexpr int total = N * C * Ho * Wo;
  const int t = blockIdx.x * blockDim.x + threadIdx.x;
  if (t >= total) return;
  const int x = t % Wo;
  const int y = (t / Wo) % Ho;
  const int c = (t / (Wo * Ho)) % C;
  const int n = t / (Wo * Ho * C);
  float fy = (y + 0.5f) * ((float)Hi / (float)Ho) - 0.5f;
  float fx = (x + 0.5f) * ((float)Wi / (float)Wo) - 0.5f;
  fy = fminf(fmaxf(fy, 0.f), (float)(Hi - 1));
  fx = fminf(fmaxf(fx, 0.f), (float)(Wi - 1));
  const int y0 = (int)floorf(fy), x0 = (int)floorf(fx);
  const int y1 = min(y0 + 1, Hi - 1), x1 = min(x0 + 1, Wi - 1);
  const float wy = fy - (float)y0, wx = fx - (float)x0;
  const float* base = in + ((size_t)(n * C + c) * Hi) * Wi;
  const float v00 = base[y0 * Wi + x0], v01 = base[y0 * Wi + x1];
  const float v10 = base[y1 * Wi + x0], v11 = base[y1 * Wi + x1];
  const float v0 = v00 + wx * (v01 - v00);
  const float v1 = v10 + wx * (v11 - v10);
  out[t] = v0 + wy * (v1 - v0);
}

// zq gather + squared error + code counts
template <int S>
__global__ __launch_bounds__(256) void vq_gather(
    const float* __restrict__ Fs, const float* __restrict__ cb,
    const int* __restrict__ idx, float* __restrict__ zq,
    float* __restrict__ sqAcc, int* __restrict__ counts) {
  constexpr int Npos = 64 * S * S;
  constexpr int SS = S * S;
  const int t = blockIdx.x * blockDim.x + threadIdx.x;
  float local = 0.f;
  if (t < Npos * 64) {
    const int pos = t >> 6, c = t & 63;
    const int n = pos / SS;
    const int rem = pos - n * SS;
    const int y = rem / S, x = rem - y * S;
    const int code = idx[pos];
    const size_t o = ((size_t)(n * 64 + c) * S + y) * S + x;
    const float q = cb[code * 64 + c];
    const float z = Fs[o];
    zq[o] = q;  // straight-through forward value == zq
    const float dd = q - z;
    local = dd * dd;
    if (c == 0) atomicAdd(&counts[code], 1);
  }
  __shared__ float red[256];
  red[threadIdx.x] = local;
  __syncthreads();
  for (int st = 128; st > 0; st >>= 1) {
    if (threadIdx.x < st) red[threadIdx.x] += red[threadIdx.x + st];
    __syncthreads();
  }
  if (threadIdx.x == 0) atomicAdd(sqAcc, red[0]);
}

__global__ __launch_bounds__(256) void vq_finalize(
    const int* __restrict__ counts, const float* __restrict__ sqAcc,
    float* __restrict__ perp, float* __restrict__ vqTotal, int Npos) {
  __shared__ float red[256];
  float s = 0.f;
  for (int i = threadIdx.x; i < 1024; i += 256) {
    const float avg = (float)counts[i] / (float)Npos;
    s += avg * logf(avg + 1e-10f);
  }
  red[threadIdx.x] = s;
  __syncthreads();
  for (int st = 128; st > 0; st >>= 1) {
    if (threadIdx.x < st) red[threadIdx.x] += red[threadIdx.x + st];
    __syncthreads();
  }
  if (threadIdx.x == 0) {
    *perp = expf(-red[0]);
    *vqTotal += 1.25f * (sqAcc[0] / (float)(Npos * 64));  // (1 + BETA) * mse
  }
}

__global__ __launch_bounds__(256) void bce_reduce(
    const float* __restrict__ logit, const float* __restrict__ tgt,
    float* __restrict__ acc, int n) {
  float s = 0.f;
  for (int i = blockIdx.x * blockDim.x + threadIdx.x; i < n;
       i += gridDim.x * blockDim.x) {
    const float v = logit[i], t = tgt[i];
    s += fmaxf(v, 0.f) - v * t + log1pf(expf(-fabsf(v)));
  }
  __shared__ float red[256];
  red[threadIdx.x] = s;
  __syncthreads();
  for (int st = 128; st > 0; st >>= 1) {
    if (threadIdx.x < st) red[threadIdx.x] += red[threadIdx.x + st];
    __syncthreads();
  }
  if (threadIdx.x == 0) atomicAdd(acc, red[0]);
}

__global__ void finalize_out(float* __restrict__ out, const float* __restrict__ bce,
                             const float* __restrict__ vqTotal,
                             const float* __restrict__ perps) {
  if (threadIdx.x == 0) out[1920000] = bce[0] / 1920000.f + vqTotal[0];
  if (threadIdx.x < 6) out[1920001 + threadIdx.x] = perps[threadIdx.x];
}

// ---------------------------------------------------------------------------
// host side
// ---------------------------------------------------------------------------
template <int Ci, int Hi, int Wi, int Co, int Ho, int Wo, int Kk, int S, int P,
          int MODE, int RELU, int ACCUM>
static void run_conv(const float* in, const float* w, const float* bias,
                     float* out, _Float16* wpack, hipStream_t stream) {
  constexpr int Ktot = Ci * Kk * Kk;
  constexpr int NKC = (Ktot + 31) / 32;
  constexpr int NCTP = ((Co + 63) / 64) * 4;
  constexpr int tot = NCTP * NKC * 512;
  repack_weights<Ci, Co, Kk, MODE><<<(tot + 255) / 256, 256, 0, stream>>>(w, wpack);
  constexpr int M = 64 * Ho * Wo;
  dim3 g((M + 31) / 32, (Co + 63) / 64);
  conv2d_wmma<Ci, Hi, Wi, Co, Ho, Wo, Kk, S, P, MODE, RELU, ACCUM>
      <<<g, 128, 0, stream>>>(in, wpack, bias, out);
}

template <int S, int I>
static void vq_scale(const float* Flat, const float* cb, const _Float16* cbh,
                     const float* cnorm, const float* phi_w, const float* phi_b,
                     float* Fs, float* zq, float* up, float* Fhat,
                     _Float16* wpack, int* idx, int* counts, float* accum,
                     hipStream_t stream) {
  constexpr int Npos = 64 * S * S;
  const float* Fsrc = Flat;
  if constexpr (S != 25) {
    constexpr int totd = 64 * 64 * S * S;
    bilinear_resize<64, 64, 25, 25, S, S>
        <<<(totd + 255) / 256, 256, 0, stream>>>(Flat, Fs);
    Fsrc = Fs;
  }
  zero_i<<<4, 256, 0, stream>>>(counts, 1024);
  zero_f<<<1, 32, 0, stream>>>(&accum[0], 1);

  constexpr int rowTiles = (Npos + 15) / 16;
  vq_argmin_wmma<S><<<(rowTiles + 7) / 8, 256, 0, stream>>>(Fsrc, cbh, cnorm, idx);

  constexpr int totg = Npos * 64;
  vq_gather<S><<<(totg + 255) / 256, 256, 0, stream>>>(Fsrc, cb, idx, zq,
                                                       &accum[0], counts);
  vq_finalize<<<1, 256, 0, stream>>>(counts, &accum[0], &accum[3 + I], &accum[2],
                                     Npos);

  const float* upsrc = zq;
  if constexpr (S != 25) {
    constexpr int totu = 64 * 64 * 25 * 25;
    bilinear_resize<64, 64, S, S, 25, 25>
        <<<(totu + 255) / 256, 256, 0, stream>>>(zq, up);
    upsrc = up;
  }
  constexpr int ACC = (I > 0) ? 1 : 0;
  run_conv<64, 25, 25, 64, 25, 25, 3, 1, 1, 0, 1, ACC>(
      upsrc, phi_w + (size_t)I * 64 * 64 * 9, phi_b + I * 64, Fhat, wpack, stream);
}

extern "C" void kernel_launch(void* const* d_in, const int* in_sizes, int n_in,
                              void* d_out, int out_size, void* d_ws, size_t ws_size,
                              hipStream_t stream) {
  (void)in_sizes; (void)n_in; (void)out_size; (void)ws_size;
  const float* x    = (const float*)d_in[0];
  const float* e0_w = (const float*)d_in[1];  const float* e0_b = (const float*)d_in[2];
  const float* e1_w = (const float*)d_in[3];  const float* e1_b = (const float*)d_in[4];
  const float* e2_w = (const float*)d_in[5];  const float* e2_b = (const float*)d_in[6];
  const float* e3_w = (const float*)d_in[7];  const float* e3_b = (const float*)d_in[8];
  const float* d0_w = (const float*)d_in[9];  const float* d0_b = (const float*)d_in[10];
  const float* d1_w = (const float*)d_in[11]; const float* d1_b = (const float*)d_in[12];
  const float* d2_w = (const float*)d_in[13]; const float* d2_b = (const float*)d_in[14];
  const float* d3_w = (const float*)d_in[15]; const float* d3_b = (const float*)d_in[16];
  const float* phi_w = (const float*)d_in[17];
  const float* phi_b = (const float*)d_in[18];
  const float* cb    = (const float*)d_in[19];
  float* out = (float*)d_out;

  // workspace carve (floats)
  float* ws = (float*)d_ws;
  size_t off = 0;
  float* h0   = ws + off; off += 10240000;  // [64,64,50,50]  (reused as g2)
  float* h1   = ws + off; off += 20480000;  // [64,128,50,50] (reused as g1)
  float* h2   = ws + off; off += 5120000;   // [64,128,25,25] (reused as g0)
  float* Flat = ws + off; off += 2560000;   // [64,64,25,25]
  float* Fs   = ws + off; off += 2560000;
  float* zq   = ws + off; off += 2560000;
  float* up   = ws + off; off += 2560000;
  float* Fhat = ws + off; off += 2560000;
  _Float16* cbh = (_Float16*)(ws + off); off += 32768;     // f16 codebook
  _Float16* wpack = (_Float16*)(ws + off); off += 131072;  // packed weights (shared)
  int*   idx    = (int*)(ws + off); off += 40000;
  int*   counts = (int*)(ws + off); off += 1024;
  float* cnorm  = ws + off; off += 1024;
  float* accum  = ws + off; off += 16;  // [0]=sq, [1]=bce, [2]=vqTotal, [3..8]=perps

  zero_f<<<1, 16, 0, stream>>>(accum, 16);
  codebook_norms<<<4, 256, 0, stream>>>(cb, cnorm);
  cast_cb_f16<<<256, 256, 0, stream>>>(cb, cbh);

  // ---------------- encoder ----------------
  run_conv<3, 100, 100, 64, 50, 50, 4, 2, 1, 0, 1, 0>(x, e0_w, e0_b, h0, wpack, stream);
  run_conv<64, 50, 50, 128, 50, 50, 3, 1, 1, 0, 1, 0>(h0, e1_w, e1_b, h1, wpack, stream);
  run_conv<128, 50, 50, 128, 25, 25, 4, 2, 1, 0, 1, 0>(h1, e2_w, e2_b, h2, wpack, stream);
  run_conv<128, 25, 25, 64, 25, 25, 3, 1, 1, 0, 0, 0>(h2, e3_w, e3_b, Flat, wpack, stream);

  // ---------------- multi-scale VQ ----------------
  vq_scale<1, 0>(Flat, cb, cbh, cnorm, phi_w, phi_b, Fs, zq, up, Fhat, wpack, idx, counts, accum, stream);
  vq_scale<5, 1>(Flat, cb, cbh, cnorm, phi_w, phi_b, Fs, zq, up, Fhat, wpack, idx, counts, accum, stream);
  vq_scale<10, 2>(Flat, cb, cbh, cnorm, phi_w, phi_b, Fs, zq, up, Fhat, wpack, idx, counts, accum, stream);
  vq_scale<15, 3>(Flat, cb, cbh, cnorm, phi_w, phi_b, Fs, zq, up, Fhat, wpack, idx, counts, accum, stream);
  vq_scale<20, 4>(Flat, cb, cbh, cnorm, phi_w, phi_b, Fs, zq, up, Fhat, wpack, idx, counts, accum, stream);
  vq_scale<25, 5>(Flat, cb, cbh, cnorm, phi_w, phi_b, Fs, zq, up, Fhat, wpack, idx, counts, accum, stream);

  // ---------------- decoder (reuse encoder buffers) ----------------
  float* g0 = h2;  // [64,128,25,25]
  float* g1 = h1;  // [64,128,50,50]
  float* g2 = h0;  // [64,64,50,50]
  run_conv<64, 25, 25, 128, 25, 25, 3, 1, 1, 0, 1, 0>(Fhat, d0_w, d0_b, g0, wpack, stream);
  run_conv<128, 25, 25, 128, 50, 50, 4, 2, 1, 1, 1, 0>(g0, d1_w, d1_b, g1, wpack, stream);
  run_conv<128, 50, 50, 64, 50, 50, 3, 1, 1, 1, 1, 0>(g1, d2_w, d2_b, g2, wpack, stream);
  run_conv<64, 50, 50, 3, 100, 100, 4, 2, 1, 1, 0, 0>(g2, d3_w, d3_b, out, wpack, stream);

  // ---------------- losses ----------------
  bce_reduce<<<2048, 256, 0, stream>>>(out, x, &accum[1], 1920000);
  finalize_out<<<1, 32, 0, stream>>>(out, &accum[1], &accum[2], &accum[3]);
}
